// AttentionPool2D_37503654429264
// MI455X (gfx1250) — matrix operand order repaired and checked by reading, not compiled, not run on hardware
//
#include <hip/hip_runtime.h>
#include <math.h>

// ---------------------------------------------------------------------------
// AttentionPool2D for MI455X (gfx1250), wave32 + WMMA f32_16x16x32_f16.
//
// Algebraic restructure: only query token l=0 survives to the output, so
//   q0[b,:]   = W_q xf[b,:,0] + b_q
//   r[b,h,:]  = W_k,h^T q0_h                      (K projection eliminated)
//   score     = (r . xf_s) * ch^-0.5              (k-bias cancels in softmax)
//   m[b,h,:]  = sum_s softmax(score)_s * xf[b,:,s]
//   a0[b,c]   = W_v[c,:] . m[b,h(c),:] + b_v[c]   (V projection eliminated)
//   out       = W_c a0 + b_c
// ~2 GFLOP total, memory bound (~81 MB HBM; xf stays in the 192 MB L2
// between passes -> ~3.5us floor at 23.3 TB/s). f16 WMMA, fp32 accumulate.
//
// This revision: dense xf0 side-buffer (kills the 576B-stride A reads in the
// q0 GEMM), paired N-tiles sharing one A fragment (2 wmma per K step, halved
// A traffic), and global_prefetch on the streamed B operand.
// ---------------------------------------------------------------------------

typedef __attribute__((ext_vector_type(16))) _Float16 v16h;
typedef __attribute__((ext_vector_type(8)))  float    v8f;

#define B_   64
#define C_   1024
#define P_   256          // S*S pixels
#define L_   257          // sequence length (mean token + pixels)
#define LP_  288          // padded: 9*32 (K steps), 18*16 (N tiles)
#define NH_  16
#define CH_  64
#define OUT_ 1024

// ---- WMMA fragment loaders (contiguous k = 16*(lane>>4) + elem order, wave32) -------------------
// A 16x32 f16: lane<16 -> elems 0..15 = K 0..15 (M = lane&15)
//              lane>=16 -> K 16..31
template <typename T>
__device__ __forceinline__ v16h load_a(const T* A, long rs, long cs, int lane) {
  const int m  = lane & 15;
  const int kb = (lane >> 4) << 4;           // 0 or 16
  const T* base = A + (long)m * rs;
  v16h a;
#pragma unroll
  for (int j = 0; j < 16; ++j) a[j] = (_Float16)(float)base[(long)(kb + j) * cs];
  return a;
}

// B 32x16 f16: lanes 0..15 hold K=0..15, lanes 16..31 hold K=16..31 (N = lane&15)
template <typename T>
__device__ __forceinline__ v16h load_b(const T* Bm, long rs, long cs, int lane) {
  const int n  = lane & 15;
  const int kb = (lane >> 4) << 4;           // 0 or 16
  const T* base = Bm + (long)n * cs + (long)kb * rs;
  v16h b;
#pragma unroll
  for (int j = 0; j < 16; ++j) b[j] = (_Float16)(float)base[(long)j * rs];
  return b;
}

// Two 16x16 output tiles per wave sharing the A fragment; B tiles at column
// offsets 0 and 16.  A(m,k)=A[m*ars+k*acs], B(k,n)=B[k*brs+n*bcs].
template <typename TA, typename TB>
__device__ __forceinline__ void gemm_tile2(const TA* A, long ars, long acs,
                                           const TB* Bm, long brs, long bcs,
                                           int K, v8f& acc0, v8f& acc1) {
  const int  lane  = threadIdx.x & 31;
  const long bstep = 16L * bcs;
  for (int k0 = 0; k0 < K; k0 += 32) {
    // Pull the next K-slab of the streamed operands toward the WGP while the
    // current WMMAs run (lowers to global_prefetch_b8 / flat_prefetch_b8).
    __builtin_prefetch((const void*)(Bm + (long)(k0 + 32) * brs), 0, 3);
    __builtin_prefetch((const void*)(A  + (long)(k0 + 32) * acs), 0, 3);
    v16h a  = load_a(A + (long)k0 * acs, ars, acs, lane);
    v16h b0 = load_b(Bm + (long)k0 * brs, brs, bcs, lane);
    v16h b1 = load_b(Bm + (long)k0 * brs + bstep, brs, bcs, lane);
    acc0 = __builtin_amdgcn_wmma_f32_16x16x32_f16(false, a, false, b0,
                                                  (short)0, acc0, false, false);
    acc1 = __builtin_amdgcn_wmma_f32_16x16x32_f16(false, a, false, b1,
                                                  (short)0, acc1, false, false);
  }
}

// C/D f32 16x16: VGPR v -> row = v + (lane>>4)*8, col = lane&15
#define TILE_STORE_PROLOG                       \
  const int lane = threadIdx.x & 31;            \
  const int col  = lane & 15;                   \
  const int rb   = (lane >> 4) << 3;

// ---- Kernel 1: xf[b,c,:] = [mean(x_row), x_row] + pos_emb, f16, zero-pad --
// Also writes the dense mean-token column xf0[b*C+c] so the q0 GEMM reads a
// unit-stride A instead of one f16 every 576 bytes.
__global__ void k_build_xf(const float* __restrict__ x, const float* __restrict__ pos,
                           _Float16* __restrict__ xf, _Float16* __restrict__ xf0) {
  const int wave = threadIdx.x >> 5;
  const int lane = threadIdx.x & 31;
  const int row  = blockIdx.x * 8 + wave;            // b*C + c
  const int c    = row & (C_ - 1);
  const float4* xr = (const float4*)(x + (size_t)row * P_);
  float4 v0 = xr[lane * 2 + 0];
  float4 v1 = xr[lane * 2 + 1];
  float s = v0.x + v0.y + v0.z + v0.w + v1.x + v1.y + v1.z + v1.w;
#pragma unroll
  for (int off = 16; off; off >>= 1) s += __shfl_xor(s, off, 32);
  const float mean = s * (1.0f / P_);
  _Float16* out = xf + (size_t)row * LP_;
  const float* pe = pos + (size_t)c * L_;
  float px[8] = {v0.x, v0.y, v0.z, v0.w, v1.x, v1.y, v1.z, v1.w};
#pragma unroll
  for (int j = 0; j < 8; ++j) {
    int l = lane * 8 + j + 1;
    out[l] = (_Float16)(px[j] + pe[l]);
  }
  if (lane == 0) {
    _Float16 t0 = (_Float16)(mean + pe[0]);
    out[0]   = t0;
    xf0[row] = t0;
  }
  if (lane < (LP_ - L_)) out[L_ + lane] = (_Float16)0.0f;   // 31 pad entries
}

// ---- Kernel 2: q0[b,o] = W_q xf0[b,:] + b_q   (M=64,N=1024,K=1024) --------
__global__ void k_q0(const _Float16* __restrict__ xf0, const float* __restrict__ wqkv,
                     const float* __restrict__ bqkv, _Float16* __restrict__ q0) {
  const int t  = blockIdx.x * 8 + (threadIdx.x >> 5);   // 128 tile-pairs
  const int tm = t >> 5, tn0 = (t & 31) * 2;
  v8f a0v = {}, a1v = {};
  gemm_tile2(xf0 + (size_t)(tm * 16) * C_, (long)C_, 1L,
             wqkv + (size_t)(tn0 * 16) * C_, 1L, (long)C_, C_, a0v, a1v);
  TILE_STORE_PROLOG
#pragma unroll
  for (int v = 0; v < 8; ++v) {
    int o0 = tn0 * 16 + col;
    size_t rowoff = (size_t)(tm * 16 + rb + v) * C_;
    q0[rowoff + o0]      = (_Float16)(a0v[v] + bqkv[o0]);
    q0[rowoff + o0 + 16] = (_Float16)(a1v[v] + bqkv[o0 + 16]);
  }
}

// ---- Kernel 3: r[b,h,:] = q0_h . W_k,h   (per head: M=64,N=1024,K=64) -----
__global__ void k_r(const _Float16* __restrict__ q0, const float* __restrict__ wqkv,
                    _Float16* __restrict__ r) {
  const int t  = blockIdx.x * 8 + (threadIdx.x >> 5);   // 2048 tile-pairs
  const int h  = t >> 7;
  const int t2 = t & 127;
  const int tm = t2 >> 5, tn0 = (t2 & 31) * 2;
  v8f a0v = {}, a1v = {};
  gemm_tile2(q0 + (size_t)(tm * 16) * C_ + h * CH_, (long)C_, 1L,
             wqkv + (size_t)(C_ + h * CH_) * C_ + tn0 * 16, (long)C_, 1L, CH_,
             a0v, a1v);
  TILE_STORE_PROLOG
#pragma unroll
  for (int v = 0; v < 8; ++v) {
    size_t rowoff = (size_t)((tm * 16 + rb + v) * NH_ + h) * C_ + tn0 * 16 + col;
    r[rowoff]      = (_Float16)a0v[v];
    r[rowoff + 16] = (_Float16)a1v[v];
  }
}

// ---- Kernel 4: sc[b,h,s] = (r_b . xf_b) * ch^-0.5  (per b: 16 x 288, K=1024)
__global__ void k_scores(const _Float16* __restrict__ r, const _Float16* __restrict__ xf,
                         float* __restrict__ sc) {
  const int t   = blockIdx.x * 8 + (threadIdx.x >> 5);  // 576 tile-pairs
  const int b   = t / 9, tn0 = (t % 9) * 2;
  v8f a0v = {}, a1v = {};
  gemm_tile2(r + (size_t)b * NH_ * C_, (long)C_, 1L,
             xf + (size_t)b * C_ * LP_ + tn0 * 16, (long)LP_, 1L, C_, a0v, a1v);
  TILE_STORE_PROLOG
#pragma unroll
  for (int v = 0; v < 8; ++v) {
    size_t rowoff = (size_t)(b * NH_ + rb + v) * LP_ + tn0 * 16 + col;
    sc[rowoff]      = a0v[v] * 0.125f;   // 1/sqrt(64)
    sc[rowoff + 16] = a1v[v] * 0.125f;
  }
}

// ---- Kernel 5: masked softmax over s<257; writes f16 weights, zero padding
__global__ void k_softmax(const float* __restrict__ sc, _Float16* __restrict__ w) {
  const int idx  = blockIdx.x * 8 + (threadIdx.x >> 5);   // b*NH + h
  const int lane = threadIdx.x & 31;
  const float* p = sc + (size_t)idx * LP_;
  float vals[9];
#pragma unroll
  for (int j = 0; j < 9; ++j) {
    int s = lane + 32 * j;
    vals[j] = (s < L_) ? p[s] : -3.0e38f;
  }
  float mx = vals[0];
#pragma unroll
  for (int j = 1; j < 9; ++j) mx = fmaxf(mx, vals[j]);
#pragma unroll
  for (int off = 16; off; off >>= 1) mx = fmaxf(mx, __shfl_xor(mx, off, 32));
  float e[9], sum = 0.0f;
#pragma unroll
  for (int j = 0; j < 9; ++j) { e[j] = __expf(vals[j] - mx); sum += e[j]; }
#pragma unroll
  for (int off = 16; off; off >>= 1) sum += __shfl_xor(sum, off, 32);
  const float inv = 1.0f / sum;
  _Float16* o = w + (size_t)idx * LP_;
#pragma unroll
  for (int j = 0; j < 9; ++j) {
    int s = lane + 32 * j;
    o[s] = (_Float16)((s < L_) ? e[j] * inv : 0.0f);
  }
}

// ---- Kernel 6: m[b,h,c'] = sum_s w[b,h,s] * xf[b,c',s]  (per b: 16x1024,K=288)
__global__ void k_m(const _Float16* __restrict__ w, const _Float16* __restrict__ xf,
                    _Float16* __restrict__ m) {
  const int t   = blockIdx.x * 8 + (threadIdx.x >> 5);  // 2048 tile-pairs
  const int b   = t >> 5, tn0 = (t & 31) * 2;
  v8f a0v = {}, a1v = {};
  gemm_tile2(w + (size_t)b * NH_ * LP_, (long)LP_, 1L,
             xf + (size_t)(b * C_ + tn0 * 16) * LP_, 1L, (long)LP_, LP_, a0v, a1v);
  TILE_STORE_PROLOG
#pragma unroll
  for (int v = 0; v < 8; ++v) {
    size_t rowoff = (size_t)(b * NH_ + rb + v) * C_ + tn0 * 16 + col;
    m[rowoff]      = (_Float16)a0v[v];
    m[rowoff + 16] = (_Float16)a1v[v];
  }
}

// ---- Kernel 7: a0[b, h*64+j] = W_v,h[j,:] . m[b,h,:] + b_v  (per h: 64x64,K=1024)
__global__ void k_a0(const _Float16* __restrict__ m, const float* __restrict__ wqkv,
                     const float* __restrict__ bqkv, _Float16* __restrict__ a0) {
  const int t  = blockIdx.x * 8 + (threadIdx.x >> 5);   // 128 tile-pairs
  const int h  = t >> 3;
  const int t2 = t & 7;
  const int tm = t2 >> 1, tn0 = (t2 & 1) * 2;
  v8f a0v = {}, a1v = {};
  gemm_tile2(m + (size_t)(tm * 16) * NH_ * C_ + (size_t)h * C_, (long)NH_ * C_, 1L,
             wqkv + (size_t)(2 * C_ + h * CH_ + tn0 * 16) * C_, 1L, (long)C_, C_,
             a0v, a1v);
  TILE_STORE_PROLOG
#pragma unroll
  for (int v = 0; v < 8; ++v) {
    int o0 = h * CH_ + tn0 * 16 + col;
    size_t rowoff = (size_t)(tm * 16 + rb + v) * C_;
    a0[rowoff + o0]      = (_Float16)(a0v[v] + bqkv[2 * C_ + o0]);
    a0[rowoff + o0 + 16] = (_Float16)(a1v[v] + bqkv[2 * C_ + o0 + 16]);
  }
}

// ---- Kernel 8: out = W_c a0 + b_c  (M=64,N=1024,K=1024), fp32 output ------
__global__ void k_out(const _Float16* __restrict__ a0, const float* __restrict__ wc,
                      const float* __restrict__ bc, float* __restrict__ out) {
  const int t  = blockIdx.x * 8 + (threadIdx.x >> 5);   // 128 tile-pairs
  const int tm = t >> 5, tn0 = (t & 31) * 2;
  v8f a0v = {}, a1v = {};
  gemm_tile2(a0 + (size_t)(tm * 16) * C_, (long)C_, 1L,
             wc + (size_t)(tn0 * 16) * C_, 1L, (long)C_, C_, a0v, a1v);
  TILE_STORE_PROLOG
#pragma unroll
  for (int v = 0; v < 8; ++v) {
    int o0 = tn0 * 16 + col;
    size_t rowoff = (size_t)(tm * 16 + rb + v) * OUT_;
    out[rowoff + o0]      = a0v[v] + bc[o0];
    out[rowoff + o0 + 16] = a1v[v] + bc[o0 + 16];
  }
}

// ---------------------------------------------------------------------------
extern "C" void kernel_launch(void* const* d_in, const int* in_sizes, int n_in,
                              void* d_out, int out_size, void* d_ws, size_t ws_size,
                              hipStream_t stream) {
  const float* x    = (const float*)d_in[0];
  const float* pos  = (const float*)d_in[1];
  const float* wqkv = (const float*)d_in[2];
  const float* bqkv = (const float*)d_in[3];
  const float* wc   = (const float*)d_in[4];
  const float* bc   = (const float*)d_in[5];
  float* out = (float*)d_out;

  char* w = (char*)d_ws;
  size_t o = 0;
  _Float16* xf  = (_Float16*)(w + o); o += (size_t)B_ * C_  * LP_ * 2;  // 37.75 MB
  _Float16* xf0 = (_Float16*)(w + o); o += (size_t)B_ * C_  * 2;        // 128 KB
  _Float16* q0  = (_Float16*)(w + o); o += (size_t)B_ * C_  * 2;        // 128 KB
  _Float16* r   = (_Float16*)(w + o); o += (size_t)B_ * NH_ * C_ * 2;   // 2 MB
  float*    sc  = (float*)   (w + o); o += (size_t)B_ * NH_ * LP_ * 4;  // 1.13 MB
  _Float16* wgt = (_Float16*)(w + o); o += (size_t)B_ * NH_ * LP_ * 2;  // 0.56 MB
  _Float16* mm  = (_Float16*)(w + o); o += (size_t)B_ * NH_ * C_ * 2;   // 2 MB
  _Float16* a0  = (_Float16*)(w + o); o += (size_t)B_ * C_  * 2;        // 128 KB

  const dim3 blk(256);
  k_build_xf<<<dim3(B_ * C_ / 8), blk, 0, stream>>>(x, pos, xf, xf0);
  k_q0     <<<dim3(16),  blk, 0, stream>>>(xf0, wqkv, bqkv, q0);
  k_r      <<<dim3(256), blk, 0, stream>>>(q0, wqkv, r);
  k_scores <<<dim3(72),  blk, 0, stream>>>(r, xf, sc);
  k_softmax<<<dim3(128), blk, 0, stream>>>(sc, wgt);
  k_m      <<<dim3(256), blk, 0, stream>>>(wgt, xf, mm);
  k_a0     <<<dim3(16),  blk, 0, stream>>>(mm, wqkv, bqkv, a0);
  k_out    <<<dim3(16),  blk, 0, stream>>>(a0, wc, bc, out);
}
